// UnidirectionalAttention_17867063951831
// MI455X (gfx1250) — compile-verified, matrix operand order checked by reading, not run on hardware
//
#include <hip/hip_runtime.h>

typedef __attribute__((ext_vector_type(16))) __bf16 v16bf;
typedef __attribute__((ext_vector_type(8)))  __bf16 v8bf;
typedef __attribute__((ext_vector_type(8)))  float  v8f;

#define WMMA_BF16(a, b, c) \
  __builtin_amdgcn_wmma_f32_16x16x32_bf16(false, (a), false, (b), (short)0, (c), false, false)

namespace {
constexpr int kB  = 2;
constexpr int kS  = 2048;
constexpr int kH  = 1024;
constexpr int kNH = 16;
constexpr int kHD = 64;
constexpr int kM  = kB * kS;   // 4096 rows total
constexpr int kN3 = 3 * kH;    // 3072
}

// Load a 16-element bf16 WMMA operand fragment (A-style / B-style identical):
// element e -> K = (e<8 ? 8*hf + e : 16 + 8*hf + (e-8)) relative to p.
static __device__ __forceinline__ v16bf ldfrag(const __bf16* p, int hf) {
  v8bf lo = *reinterpret_cast<const v8bf*>(p + 8 * hf);
  v8bf hi = *reinterpret_cast<const v8bf*>(p + 16 + 8 * hf);
  return __builtin_shufflevector(lo, hi, 0, 1, 2, 3, 4, 5, 6, 7,
                                 8, 9, 10, 11, 12, 13, 14, 15);
}

__global__ void cvt_bf16_kernel(const float* __restrict__ s, __bf16* __restrict__ d, int n) {
  int i = blockIdx.x * blockDim.x + threadIdx.x;
  if (i < n) d[i] = (__bf16)s[i];
}

// dst[n][k] = src[k][n]  (K x N row-major -> N x K row-major, with f32->bf16)
__global__ void tcvt_bf16_kernel(const float* __restrict__ s, __bf16* __restrict__ d,
                                 int K, int N) {
  int i = blockIdx.x * blockDim.x + threadIdx.x;
  if (i < K * N) {
    int k = i / N, n = i - k * N;
    d[(size_t)n * K + k] = (__bf16)s[i];
  }
}

// qkv = x @ w_qkv + b_qkv ; scatter into Q[b][h][s][d] (pre-scaled by 1/sqrt(hd)),
// K[b][h][s][d], V^T[b][h][d][s].  One wave computes a 64x64 output tile:
// 16 WMMAs per K-step on 8 operand fragments (1 b128 load per WMMA).
__global__ void __launch_bounds__(32)
qkv_gemm_kernel(const __bf16* __restrict__ xb,   // [4096][1024]
                const __bf16* __restrict__ wt,   // [3072][1024] = w_qkv^T
                const float*  __restrict__ bias, // [3072]
                __bf16* __restrict__ qo,
                __bf16* __restrict__ ko,
                __bf16* __restrict__ vto) {
  const int lane = threadIdx.x & 31;
  const int hf = lane >> 4, l16 = lane & 15;
  const int n0 = blockIdx.x * 64;
  const int m0 = blockIdx.y * 64;
  const __bf16* arow = xb + (size_t)(m0 + l16) * kH;
  const __bf16* brow = wt + (size_t)(n0 + l16) * kH;
  const v8f z = {0.f, 0.f, 0.f, 0.f, 0.f, 0.f, 0.f, 0.f};
  v8f c[4][4] = {{z, z, z, z}, {z, z, z, z}, {z, z, z, z}, {z, z, z, z}};
#pragma unroll 1
  for (int kk = 0; kk < kH; kk += 32) {
    v16bf a[4], b[4];
#pragma unroll
    for (int mi = 0; mi < 4; ++mi) a[mi] = ldfrag(arow + (size_t)(16 * mi) * kH + kk, hf);
#pragma unroll
    for (int ni = 0; ni < 4; ++ni) b[ni] = ldfrag(brow + (size_t)(16 * ni) * kH + kk, hf);
#pragma unroll
    for (int mi = 0; mi < 4; ++mi)
#pragma unroll
      for (int ni = 0; ni < 4; ++ni)
        c[mi][ni] = WMMA_BF16(a[mi], b[ni], c[mi][ni]);
  }
  // 64-wide N tile lies in exactly one of {q,k,v} and one head (1024 % 64 == 0).
  const int which = n0 >> 10;
  const int head = (n0 & 1023) >> 6;
#pragma unroll
  for (int ni = 0; ni < 4; ++ni) {
    const int n = n0 + 16 * ni + l16;
    const int d = n & 63;
    const float bv = bias[n];
#pragma unroll
    for (int mi = 0; mi < 4; ++mi)
#pragma unroll
      for (int r = 0; r < 8; ++r) {
        const int m = m0 + 16 * mi + r + 8 * hf;
        const int bb = m >> 11;
        const int ss = m & 2047;
        const float val = c[mi][ni][r] + bv;
        if (which == 0)
          qo[(((size_t)bb * kNH + head) * kS + ss) * kHD + d] = (__bf16)(val * 0.125f);
        else if (which == 1)
          ko[(((size_t)bb * kNH + head) * kS + ss) * kHD + d] = (__bf16)val;
        else
          vto[(((size_t)bb * kNH + head) * kHD + d) * kS + ss] = (__bf16)val;
      }
  }
}

// Flash attention, transposed formulation. One wave per 32 query rows (two
// 16-row i-tiles sharing all K/V fragments). S^T = K Q^T (C layout of S^T ==
// B-operand layout of P^T, so no shuffles/LDS), out^T = V^T P^T.
__global__ void __launch_bounds__(32)
attn_kernel(const __bf16* __restrict__ q,   // [b][h][s][64], pre-scaled
            const __bf16* __restrict__ k,   // [b][h][s][64]
            const __bf16* __restrict__ vt,  // [b][h][64][s]
            __bf16* __restrict__ ao) {      // [b*s][1024]
  const int lane = threadIdx.x & 31;
  const int hf = lane >> 4, l16 = lane & 15;
  const int i0 = blockIdx.x * 32;
  const int h = blockIdx.y;
  const int b = blockIdx.z;
  const __bf16* qh = q + ((size_t)(b * kNH + h) * kS) * kHD;
  const __bf16* kh = k + ((size_t)(b * kNH + h) * kS) * kHD;
  const __bf16* vh = vt + ((size_t)(b * kNH + h) * kHD) * kS;

  v16bf bQ[2][2];
#pragma unroll
  for (int it = 0; it < 2; ++it)
#pragma unroll
    for (int cc = 0; cc < 2; ++cc)
      bQ[it][cc] = ldfrag(qh + (size_t)(i0 + 16 * it + l16) * kHD + 32 * cc, hf);

  const v8f z = {0.f, 0.f, 0.f, 0.f, 0.f, 0.f, 0.f, 0.f};
  v8f acc[2][4] = {{z, z, z, z}, {z, z, z, z}};
  float mrun[2] = {-1e30f, -1e30f};
  float lrun[2] = {0.f, 0.f};

#pragma unroll 1
  for (int j0 = i0; j0 < kS; j0 += 32) {
    v16bf aK[2][2];
#pragma unroll
    for (int t = 0; t < 2; ++t)
#pragma unroll
      for (int cc = 0; cc < 2; ++cc)
        aK[t][cc] = ldfrag(kh + (size_t)(j0 + 16 * t + l16) * kHD + 32 * cc, hf);

    v8f st[2][2];
#pragma unroll
    for (int it = 0; it < 2; ++it)
#pragma unroll
      for (int t = 0; t < 2; ++t) {
        st[it][t] = WMMA_BF16(aK[t][0], bQ[it][0], z);
        st[it][t] = WMMA_BF16(aK[t][1], bQ[it][1], st[it][t]);
      }

    if (j0 == i0) {  // leading block: mask j < i
#pragma unroll
      for (int it = 0; it < 2; ++it) {
        const int icol = i0 + 16 * it + l16;
#pragma unroll
        for (int t = 0; t < 2; ++t)
#pragma unroll
          for (int r = 0; r < 8; ++r)
            if (j0 + 16 * t + r + 8 * hf < icol) st[it][t][r] = -1e30f;
      }
    }

    v16bf bP[2];
#pragma unroll
    for (int it = 0; it < 2; ++it) {
      float mx = -1e30f;
#pragma unroll
      for (int t = 0; t < 2; ++t)
#pragma unroll
        for (int r = 0; r < 8; ++r) mx = fmaxf(mx, st[it][t][r]);
      mx = fmaxf(mx, __shfl_xor(mx, 16, 32));
      const float mnew = fmaxf(mrun[it], mx);
      float p[2][8], rs = 0.f;
#pragma unroll
      for (int t = 0; t < 2; ++t)
#pragma unroll
        for (int r = 0; r < 8; ++r) {
          p[t][r] = __expf(st[it][t][r] - mnew);
          rs += p[t][r];
        }
      rs += __shfl_xor(rs, 16, 32);
      const float alpha = __expf(mrun[it] - mnew);
      lrun[it] = lrun[it] * alpha + rs;
      mrun[it] = mnew;
#pragma unroll
      for (int f = 0; f < 4; ++f)
#pragma unroll
        for (int r = 0; r < 8; ++r) acc[it][f][r] *= alpha;
#pragma unroll
      for (int e = 0; e < 8; ++e) {
        bP[it][e] = (__bf16)p[0][e];
        bP[it][e + 8] = (__bf16)p[1][e];
      }
    }

#pragma unroll
    for (int f = 0; f < 4; ++f) {
      const v16bf aV = ldfrag(vh + (size_t)(16 * f + l16) * kS + j0, hf);
#pragma unroll
      for (int it = 0; it < 2; ++it)
        acc[it][f] = WMMA_BF16(aV, bP[it], acc[it][f]);
    }
  }

#pragma unroll
  for (int it = 0; it < 2; ++it) {
    const float inv = 1.f / lrun[it];  // diagonal always unmasked -> lrun > 0
    __bf16* orow = ao + ((size_t)(b * kS + i0 + 16 * it + l16)) * kH + h * kHD;
#pragma unroll
    for (int f = 0; f < 4; ++f)
#pragma unroll
      for (int r = 0; r < 8; ++r)
        orow[16 * f + r + 8 * hf] = (__bf16)(acc[it][f][r] * inv);
  }
}

// out = attn @ w_out + b_out (f32 result to d_out). 64x64 tile per wave.
__global__ void __launch_bounds__(32)
out_gemm_kernel(const __bf16* __restrict__ ab,   // [4096][1024]
                const __bf16* __restrict__ wot,  // [1024][1024] = w_out^T
                const float*  __restrict__ bias, // [1024]
                float* __restrict__ out) {
  const int lane = threadIdx.x & 31;
  const int hf = lane >> 4, l16 = lane & 15;
  const int n0 = blockIdx.x * 64;
  const int m0 = blockIdx.y * 64;
  const __bf16* arow = ab + (size_t)(m0 + l16) * kH;
  const __bf16* brow = wot + (size_t)(n0 + l16) * kH;
  const v8f z = {0.f, 0.f, 0.f, 0.f, 0.f, 0.f, 0.f, 0.f};
  v8f c[4][4] = {{z, z, z, z}, {z, z, z, z}, {z, z, z, z}, {z, z, z, z}};
#pragma unroll 1
  for (int kk = 0; kk < kH; kk += 32) {
    v16bf a[4], b[4];
#pragma unroll
    for (int mi = 0; mi < 4; ++mi) a[mi] = ldfrag(arow + (size_t)(16 * mi) * kH + kk, hf);
#pragma unroll
    for (int ni = 0; ni < 4; ++ni) b[ni] = ldfrag(brow + (size_t)(16 * ni) * kH + kk, hf);
#pragma unroll
    for (int mi = 0; mi < 4; ++mi)
#pragma unroll
      for (int ni = 0; ni < 4; ++ni)
        c[mi][ni] = WMMA_BF16(a[mi], b[ni], c[mi][ni]);
  }
#pragma unroll
  for (int ni = 0; ni < 4; ++ni) {
    const int n = n0 + 16 * ni + l16;
    const float bv = bias[n];
#pragma unroll
    for (int mi = 0; mi < 4; ++mi)
#pragma unroll
      for (int r = 0; r < 8; ++r)
        out[(size_t)(m0 + 16 * mi + r + 8 * hf) * kH + n] = c[mi][ni][r] + bv;
  }
}

extern "C" void kernel_launch(void* const* d_in, const int* in_sizes, int n_in,
                              void* d_out, int out_size, void* d_ws, size_t ws_size,
                              hipStream_t stream) {
  (void)in_sizes; (void)n_in; (void)out_size; (void)ws_size;
  const float* x      = (const float*)d_in[0];
  const float* w_qkv  = (const float*)d_in[1];
  const float* b_qkv  = (const float*)d_in[2];
  const float* w_out  = (const float*)d_in[3];
  const float* b_out  = (const float*)d_in[4];
  float* out = (float*)d_out;

  char* ws = (char*)d_ws;
  auto carve = [&](size_t bytes) {
    char* p = ws;
    ws += (bytes + 255) & ~(size_t)255;
    return p;
  };
  __bf16* xb  = (__bf16*)carve((size_t)kM * kH * 2);              // x in bf16
  __bf16* wqt = (__bf16*)carve((size_t)kN3 * kH * 2);             // w_qkv^T bf16
  __bf16* wot = (__bf16*)carve((size_t)kH * kH * 2);              // w_out^T bf16
  __bf16* qb  = (__bf16*)carve((size_t)kB * kNH * kS * kHD * 2);  // Q (pre-scaled)
  __bf16* kb  = (__bf16*)carve((size_t)kB * kNH * kS * kHD * 2);  // K
  __bf16* vtb = (__bf16*)carve((size_t)kB * kNH * kS * kHD * 2);  // V^T
  __bf16* atb = (__bf16*)carve((size_t)kM * kH * 2);              // attention out

  {
    int n = kM * kH;
    cvt_bf16_kernel<<<(n + 255) / 256, 256, 0, stream>>>(x, xb, n);
  }
  {
    int n = kH * kN3;
    tcvt_bf16_kernel<<<(n + 255) / 256, 256, 0, stream>>>(w_qkv, wqt, kH, kN3);
  }
  {
    int n = kH * kH;
    tcvt_bf16_kernel<<<(n + 255) / 256, 256, 0, stream>>>(w_out, wot, kH, kH);
  }
  qkv_gemm_kernel<<<dim3(kN3 / 64, kM / 64), 32, 0, stream>>>(xb, wqt, b_qkv, qb, kb, vtb);
  attn_kernel<<<dim3(kS / 32, kNH, kB), 32, 0, stream>>>(qb, kb, vtb, atb);
  out_gemm_kernel<<<dim3(kH / 64, kM / 64), 32, 0, stream>>>(atb, wot, b_out, out);
}
